// E_GCL_88227218194812
// MI455X (gfx1250) — matrix operand order repaired and checked by reading, not compiled, not run on hardware
//
#include <hip/hip_runtime.h>
#include <hip/hip_bf16.h>

#define DD 128
#define HH 128

typedef __attribute__((ext_vector_type(16))) _Float16 v16h;
typedef __attribute__((ext_vector_type(8)))  _Float16 v8h;
typedef __attribute__((ext_vector_type(8)))  float    v8f;

__device__ __forceinline__ float silu_f(float v) {
  // fast: x * rcp(1 + e^-x)  (single v_rcp_f32 instead of IEEE div sequence)
  return v * __builtin_amdgcn_rcpf(1.0f + __expf(-v));
}

__device__ __forceinline__ void atomic_add_f32(float* p, float v) {
  // agent-scope relaxed fadd -> global_atomic_add_f32 (no CAS loop)
  __hip_atomic_fetch_add(p, v, __ATOMIC_RELAXED, __HIP_MEMORY_SCOPE_AGENT);
}

__device__ __forceinline__ v8f wmma_f16(v16h a, v16h b, v8f c) {
  // D = A(16x32 f16) x B(32x16 f16) + C(16x16 f32)
  return __builtin_amdgcn_wmma_f32_16x16x32_f16(false, a, false, b, (short)0, c, false, false);
}

// A-operand loader: 16-bit A 16x32 layout (ISA 7.12.2):
//   lane L (M = L&15):  halves 0..7  -> K = kg*8 + 0..7
//                       halves 8..15 -> K = 16 + kg*8 + 0..7   (kg = L>>4)
// rowbase points at element [row][ks*32] of a row-major f16 matrix.
__device__ __forceinline__ v16h load_a16(const _Float16* rowbase, int kg8) {
  v8h lo = *(const v8h*)(rowbase + kg8);
  v8h hi = *(const v8h*)(rowbase + 16 + kg8);
  v16h a;
#pragma unroll
  for (int i = 0; i < 8; ++i) { a[i] = lo[i]; a[i + 8] = hi[i]; }
  return a;
}

// Same A layout but converting from an f32 row-major source (agg buffer).
__device__ __forceinline__ v16h load_a32(const float* rowbase, int kg8) {
  v16h a;
#pragma unroll
  for (int i = 0; i < 8; ++i) {
    a[i]     = (_Float16)rowbase[kg8 + i];
    a[i + 8] = (_Float16)rowbase[16 + kg8 + i];
  }
  return a;
}

// B-operand loader from pre-swizzled pack: tile (ks,nt), per-lane 16 contiguous halves.
__device__ __forceinline__ v16h load_b(const _Float16* pack, int ks, int nt, int lane) {
  return *(const v16h*)(pack + (((ks * 8 + nt) * 32 + lane) << 4));
}

// ---------------- prep kernels ----------------

__global__ void zero_f32_kernel(float* p, long long n) {
  long long i = (long long)blockIdx.x * blockDim.x + threadIdx.x;
  long long stride = (long long)gridDim.x * blockDim.x;
  for (; i < n; i += stride) p[i] = 0.0f;
}

__global__ void cvt_f16_kernel(const float* __restrict__ s, _Float16* __restrict__ d, long long n) {
  long long i = (long long)blockIdx.x * blockDim.x + threadIdx.x;
  long long stride = (long long)gridDim.x * blockDim.x;
  for (; i < n; i += stride) d[i] = (_Float16)s[i];
}

// Swizzle a 128x128 f32 weight (row-major, [K][N]) into WMMA-B tile order:
// dst[((ks*8+nt)*32 + lane)*16 + h] = W[ks*32 + (h>>3)*16 + (lane>>4)*8 + (h&7)][nt*16 + (lane&15)]
__global__ void pack_w_kernel(const float* __restrict__ src, _Float16* __restrict__ dst) {
  int idx = blockIdx.x * blockDim.x + threadIdx.x;
  if (idx >= 128 * 128) return;
  int h  = idx & 15;
  int l  = (idx >> 4) & 31;
  int t  = idx >> 9;
  int nt = t & 7;
  int ks = t >> 3;
  int n = nt * 16 + (l & 15);
  int k = ks * 32 + (h >> 3) * 16 + (l >> 4) * 8 + (h & 7);
  dst[idx] = (_Float16)src[k * 128 + n];
}

// ---------------- fused edge kernel ----------------
// One wave32 per 16-edge tile; 8 waves / 256-thread block.
__global__ __launch_bounds__(256) void egcl_edge_kernel(
    const _Float16* __restrict__ xh, const long long* __restrict__ ei,
    const float* __restrict__ coord,
    const _Float16* __restrict__ pWe1a, const _Float16* __restrict__ pWe1b,
    const float* __restrict__ we1c, const float* __restrict__ be1,
    const _Float16* __restrict__ pWe2, const float* __restrict__ be2,
    const _Float16* __restrict__ pWc1, const float* __restrict__ bc1,
    const float* __restrict__ wc2,
    float* __restrict__ agg, float* __restrict__ aggc, float* __restrict__ cnt,
    int Ecnt)
{
  __shared__ __align__(16) _Float16 smA[8][16 * HH];
  __shared__ float smRad[8][16];
  __shared__ int   smRow[8][16];

  const int lane = threadIdx.x & 31;
  const int wv   = threadIdx.x >> 5;
  const int m    = lane & 15;
  const int kg8  = (lane >> 4) * 8;

  const int tile  = blockIdx.x * 8 + wv;
  const int ebase = tile * 16;
  const bool active = (ebase < Ecnt);
  int e = ebase + m;
  if (e >= Ecnt) e = Ecnt - 1;

  const long long r = ei[e];
  const long long c = ei[(long long)Ecnt + e];

  const float dx = coord[r * 3 + 0] - coord[c * 3 + 0];
  const float dy = coord[r * 3 + 1] - coord[c * 3 + 1];
  const float dz = coord[r * 3 + 2] - coord[c * 3 + 2];
  const float rad = dx * dx + dy * dy + dz * dz;
  if (lane < 16) { smRad[wv][m] = rad; smRow[wv][m] = (int)r; }

  // Gather A operands: x[row] and x[col] rows (f16), K = 0..127 in four 32-wide steps.
  v16h Ar[4], Acl[4];
#pragma unroll
  for (int ks = 0; ks < 4; ++ks) {
    Ar[ks]  = load_a16(xh + (size_t)r * DD + ks * 32, kg8);
    Acl[ks] = load_a16(xh + (size_t)c * DD + ks * 32, kg8);
  }
  __syncthreads();

  v8f acc[8];

  // ---- GEMM1: h1 = silu(xr@We1a + xc@We1b + radial*we1c + be1) -> smA (f16) ----
  // ks-outer / nt-inner: 8 independent accumulators per K-step -> loads overlap WMMAs.
#pragma unroll
  for (int nt = 0; nt < 8; ++nt) {
    const int ncol = nt * 16 + m;
    const float b0 = be1[ncol], wr = we1c[ncol];
#pragma unroll
    for (int v = 0; v < 8; ++v) acc[nt][v] = b0 + smRad[wv][kg8 + v] * wr;
  }
#pragma unroll
  for (int ks = 0; ks < 4; ++ks)
#pragma unroll
    for (int nt = 0; nt < 8; ++nt)
      acc[nt] = wmma_f16(Ar[ks], load_b(pWe1a, ks, nt, lane), acc[nt]);
#pragma unroll
  for (int ks = 0; ks < 4; ++ks)
#pragma unroll
    for (int nt = 0; nt < 8; ++nt)
      acc[nt] = wmma_f16(Acl[ks], load_b(pWe1b, ks, nt, lane), acc[nt]);
#pragma unroll
  for (int nt = 0; nt < 8; ++nt)
#pragma unroll
    for (int v = 0; v < 8; ++v)
      smA[wv][(kg8 + v) * HH + nt * 16 + m] = (_Float16)silu_f(acc[nt][v]);
  __syncthreads();

  // ---- GEMM2: edge_feat = silu(h1@We2 + be2) -> smA (overwrites h1) ----
  v16h A2[4];
#pragma unroll
  for (int ks = 0; ks < 4; ++ks) A2[ks] = load_a16(&smA[wv][m * HH + ks * 32], kg8);
#pragma unroll
  for (int nt = 0; nt < 8; ++nt)
#pragma unroll
    for (int v = 0; v < 8; ++v) acc[nt][v] = be2[nt * 16 + m];
#pragma unroll
  for (int ks = 0; ks < 4; ++ks)
#pragma unroll
    for (int nt = 0; nt < 8; ++nt)
      acc[nt] = wmma_f16(A2[ks], load_b(pWe2, ks, nt, lane), acc[nt]);
#pragma unroll
  for (int nt = 0; nt < 8; ++nt)
#pragma unroll
    for (int v = 0; v < 8; ++v)
      smA[wv][(kg8 + v) * HH + nt * 16 + m] = (_Float16)silu_f(acc[nt][v]);
  __syncthreads();

  // ---- GEMM3: c1 = silu(edge_feat@Wc1 + bc1) (accumulators kept in regs) ----
  v16h A3[4];
#pragma unroll
  for (int ks = 0; ks < 4; ++ks) A3[ks] = load_a16(&smA[wv][m * HH + ks * 32], kg8);
#pragma unroll
  for (int nt = 0; nt < 8; ++nt)
#pragma unroll
    for (int v = 0; v < 8; ++v) acc[nt][v] = bc1[nt * 16 + m];
#pragma unroll
  for (int ks = 0; ks < 4; ++ks)
#pragma unroll
    for (int nt = 0; nt < 8; ++nt)
      acc[nt] = wmma_f16(A3[ks], load_b(pWc1, ks, nt, lane), acc[nt]);

  // ---- scatter edge_feat into agg[row] (while smA still holds edge_feat) ----
  if (active) {
    for (int i = lane; i < 16 * HH; i += 32) {
      const int mm = i >> 7;
      const int nn = i & 127;
      const long long rr = smRow[wv][mm];
      atomic_add_f32(&agg[(size_t)rr * HH + nn], (float)smA[wv][i]);
    }
  }
  __syncthreads();

  // ---- overwrite smA with silu(c1), then per-edge dot with Wc2 and coord scatter ----
#pragma unroll
  for (int nt = 0; nt < 8; ++nt)
#pragma unroll
    for (int v = 0; v < 8; ++v)
      smA[wv][(kg8 + v) * HH + nt * 16 + m] = (_Float16)silu_f(acc[nt][v]);
  __syncthreads();

  if (active && lane < 16) {
    float wsum = 0.0f;
#pragma unroll
    for (int b8 = 0; b8 < 16; ++b8) {
      v8h ev = *(const v8h*)&smA[wv][m * HH + b8 * 8];
      const float4 w0 = *(const float4*)(wc2 + b8 * 8);
      const float4 w1 = *(const float4*)(wc2 + b8 * 8 + 4);
      wsum += (float)ev[0] * w0.x + (float)ev[1] * w0.y + (float)ev[2] * w0.z + (float)ev[3] * w0.w
            + (float)ev[4] * w1.x + (float)ev[5] * w1.y + (float)ev[6] * w1.z + (float)ev[7] * w1.w;
    }
    atomic_add_f32(&aggc[r * 3 + 0], dx * wsum);
    atomic_add_f32(&aggc[r * 3 + 1], dy * wsum);
    atomic_add_f32(&aggc[r * 3 + 2], dz * wsum);
    atomic_add_f32(&cnt[r], 1.0f);
  }
}

// ---------------- fused node kernel ----------------
__global__ __launch_bounds__(256) void egcl_node_kernel(
    const float* __restrict__ x, const _Float16* __restrict__ xh,
    const float* __restrict__ coord,
    const _Float16* __restrict__ pWn1a, const _Float16* __restrict__ pWn1b,
    const float* __restrict__ bn1,
    const _Float16* __restrict__ pWn2, const float* __restrict__ bn2,
    const float* __restrict__ agg, const float* __restrict__ aggc,
    const float* __restrict__ cnt,
    float* __restrict__ xout, float* __restrict__ cout, int Ncnt)
{
  __shared__ __align__(16) _Float16 smA[8][16 * HH];

  const int lane = threadIdx.x & 31;
  const int wv   = threadIdx.x >> 5;
  const int m    = lane & 15;
  const int kg8  = (lane >> 4) * 8;

  const int NT = (Ncnt + 15) / 16;
  int tile = blockIdx.x * 8 + wv;
  if (tile >= NT) tile = NT - 1;   // duplicate last tile: identical redundant writes
  const int nb = tile * 16;
  int node = nb + m;
  if (node >= Ncnt) node = Ncnt - 1;

  v16h A1[4], A2[4];
#pragma unroll
  for (int ks = 0; ks < 4; ++ks) {
    A1[ks] = load_a16(xh + (size_t)node * DD + ks * 32, kg8);
    A2[ks] = load_a32(agg + (size_t)node * HH + ks * 32, kg8);
  }

  v8f acc[8];

  // ---- GEMM1: h = silu([x, agg]@Wn1 + bn1) -> smA ----
#pragma unroll
  for (int nt = 0; nt < 8; ++nt)
#pragma unroll
    for (int v = 0; v < 8; ++v) acc[nt][v] = bn1[nt * 16 + m];
#pragma unroll
  for (int ks = 0; ks < 4; ++ks)
#pragma unroll
    for (int nt = 0; nt < 8; ++nt)
      acc[nt] = wmma_f16(A1[ks], load_b(pWn1a, ks, nt, lane), acc[nt]);
#pragma unroll
  for (int ks = 0; ks < 4; ++ks)
#pragma unroll
    for (int nt = 0; nt < 8; ++nt)
      acc[nt] = wmma_f16(A2[ks], load_b(pWn1b, ks, nt, lane), acc[nt]);
#pragma unroll
  for (int nt = 0; nt < 8; ++nt)
#pragma unroll
    for (int v = 0; v < 8; ++v)
      smA[wv][(kg8 + v) * HH + nt * 16 + m] = (_Float16)silu_f(acc[nt][v]);
  __syncthreads();

  // ---- GEMM2: x_out = x + h@Wn2 + bn2 ----
  v16h A3[4];
#pragma unroll
  for (int ks = 0; ks < 4; ++ks) A3[ks] = load_a16(&smA[wv][m * HH + ks * 32], kg8);
#pragma unroll
  for (int nt = 0; nt < 8; ++nt)
#pragma unroll
    for (int v = 0; v < 8; ++v) acc[nt][v] = bn2[nt * 16 + m];
#pragma unroll
  for (int ks = 0; ks < 4; ++ks)
#pragma unroll
    for (int nt = 0; nt < 8; ++nt)
      acc[nt] = wmma_f16(A3[ks], load_b(pWn2, ks, nt, lane), acc[nt]);
#pragma unroll
  for (int nt = 0; nt < 8; ++nt) {
    const int ncol = nt * 16 + m;
#pragma unroll
    for (int v = 0; v < 8; ++v) {
      int nd = nb + kg8 + v;
      if (nd >= Ncnt) nd = Ncnt - 1;
      xout[(size_t)nd * DD + ncol] = x[(size_t)nd * DD + ncol] + acc[nt][v];
    }
  }

  // ---- coord_out = coord + agg_c / max(cnt,1) ----
  if (lane < 16) {
    float cc = cnt[node];
    cc = fmaxf(cc, 1.0f);
#pragma unroll
    for (int j = 0; j < 3; ++j)
      cout[(size_t)node * 3 + j] = coord[(size_t)node * 3 + j] + aggc[(size_t)node * 3 + j] / cc;
  }
}

// ---------------- host launcher ----------------

extern "C" void kernel_launch(void* const* d_in, const int* in_sizes, int n_in,
                              void* d_out, int out_size, void* d_ws, size_t ws_size,
                              hipStream_t stream) {
  const float*     x     = (const float*)d_in[0];
  const long long* ei    = (const long long*)d_in[1];   // jnp.int64 edge_index [2,E]
  const float*     coord = (const float*)d_in[2];
  const float*     We1   = (const float*)d_in[3];   // [257,128]
  const float*     be1   = (const float*)d_in[4];
  const float*     We2   = (const float*)d_in[5];   // [128,128]
  const float*     be2   = (const float*)d_in[6];
  const float*     Wn1   = (const float*)d_in[7];   // [256,128]
  const float*     bn1   = (const float*)d_in[8];
  const float*     Wn2   = (const float*)d_in[9];   // [128,128]
  const float*     bn2   = (const float*)d_in[10];
  const float*     Wc1   = (const float*)d_in[11];  // [128,128]
  const float*     bc1   = (const float*)d_in[12];
  const float*     Wc2   = (const float*)d_in[13];  // [128,1]

  const int N = in_sizes[0] / DD;
  const int E = in_sizes[1] / 2;

  // workspace layout (bytes), 256-aligned
  char* ws = (char*)d_ws;
  size_t off = 0;
  auto take = [&](size_t bytes) { char* p = ws + off; off = (off + bytes + 255) & ~(size_t)255; return p; };
  float*     agg  = (float*)take((size_t)N * HH * 4);
  float*     aggc = (float*)take((size_t)N * 3 * 4);
  float*     cnt  = (float*)take((size_t)N * 4);
  _Float16*  xh   = (_Float16*)take((size_t)N * DD * 2);
  _Float16*  pWe1a = (_Float16*)take(128 * 128 * 2);
  _Float16*  pWe1b = (_Float16*)take(128 * 128 * 2);
  _Float16*  pWe2  = (_Float16*)take(128 * 128 * 2);
  _Float16*  pWc1  = (_Float16*)take(128 * 128 * 2);
  _Float16*  pWn1a = (_Float16*)take(128 * 128 * 2);
  _Float16*  pWn1b = (_Float16*)take(128 * 128 * 2);
  _Float16*  pWn2  = (_Float16*)take(128 * 128 * 2);
  (void)ws_size; (void)n_in; (void)out_size;

  // zero segment-sum buffers
  zero_f32_kernel<<<1024, 256, 0, stream>>>(agg,  (long long)N * HH);
  zero_f32_kernel<<<256, 256, 0, stream>>>(aggc, (long long)N * 3);
  zero_f32_kernel<<<128, 256, 0, stream>>>(cnt,  (long long)N);

  // x -> f16
  cvt_f16_kernel<<<2048, 256, 0, stream>>>(x, xh, (long long)N * DD);

  // weight swizzles (each 128x128)
  const int pg = (128 * 128 + 255) / 256;
  pack_w_kernel<<<pg, 256, 0, stream>>>(We1,              pWe1a);
  pack_w_kernel<<<pg, 256, 0, stream>>>(We1 + 128 * 128,  pWe1b);
  pack_w_kernel<<<pg, 256, 0, stream>>>(We2,              pWe2);
  pack_w_kernel<<<pg, 256, 0, stream>>>(Wc1,              pWc1);
  pack_w_kernel<<<pg, 256, 0, stream>>>(Wn1,              pWn1a);
  pack_w_kernel<<<pg, 256, 0, stream>>>(Wn1 + 128 * 128,  pWn1b);
  pack_w_kernel<<<pg, 256, 0, stream>>>(Wn2,              pWn2);

  const float* we1c = We1 + 256 * 128;  // radial row of We1

  // edge kernel: one wave per 16 edges, 8 waves per block
  const int etiles  = (E + 15) / 16;
  const int eblocks = (etiles + 7) / 8;
  egcl_edge_kernel<<<eblocks, 256, 0, stream>>>(
      xh, ei, coord, pWe1a, pWe1b, we1c, be1, pWe2, be2, pWc1, bc1, Wc2,
      agg, aggc, cnt, E);

  // node kernel
  float* xout = (float*)d_out;
  float* cout = xout + (size_t)N * DD;
  const int ntiles  = (N + 15) / 16;
  const int nblocks = (ntiles + 7) / 8;
  egcl_node_kernel<<<nblocks, 256, 0, stream>>>(
      x, xh, coord, pWn1a, pWn1b, bn1, pWn2, bn2, agg, aggc, cnt,
      xout, cout, N);
}
